// FullAttention_new_84911503442243
// MI455X (gfx1250) — matrix operand order repaired and checked by reading, not compiled, not run on hardware
//
#include <hip/hip_runtime.h>
#include <math.h>

// CDNA5 / gfx1250 fused gated-attention (flash-attention style, wave32 WMMA).
// B=4, L=S=1024, H=16, E=32, D=32.
// - Two 16-row l-tiles per wave: every B-tile load / V-stage feeds 12 WMMAs.
// - V chunk staged with GLOBAL_LOAD_ASYNC_TO_LDS_B128 (ASYNCcnt path): no
//   VGPR round trip, copy overlaps the score WMMAs + gate transcendentals.

typedef __attribute__((ext_vector_type(16))) _Float16 v16h;
typedef __attribute__((ext_vector_type(8)))  float    v8f;
typedef __attribute__((address_space(3)))    float    lds_f;

static constexpr int cB = 4, cL = 1024, cH = 16, cE = 32, cD = 32, cS = 1024;

#define WMMA16F(A, Bm, C) \
  __builtin_amdgcn_wmma_f32_16x16x32_f16(false, (A), false, (Bm), (short)0, (C), false, false)

__device__ __forceinline__ float clip1000(float x) {
  return fminf(fmaxf(x, -1000.0f), 1000.0f);
}

__global__ __launch_bounds__(32) void gated_attn_kernel(
    const float* __restrict__ q, const float* __restrict__ k,
    const float* __restrict__ v, const float* __restrict__ w,
    float* __restrict__ out) {
  // LDS scratch: two P tiles (C->A layout round trip) + V stage (padded rows)
  __shared__ float Pld[2][16 * 32];
  __shared__ float Vld[32 * 33];

  const int lane = threadIdx.x;   // 0..31 (wave32)
  const int nIdx = lane & 15;     // column N / row M depending on role
  const int hh   = lane >> 4;     // half-wave select

  const int blk = blockIdx.x;     // B*H*(L/32) = 2048 blocks
  const int lt  = blk & 31;
  const int h   = (blk >> 5) & 15;
  const int b   = blk >> 9;
  const int l0  = lt * 32;        // 32 query rows per wave (two 16-row tiles)

  const float scale = 0.17677669529663687f;  // 1/sqrt(32)

  // ---------------- A tiles (held for whole kernel) ----------------
  // 16-bit A 16x32 tile: lane(M=nIdx, hh): elems 0..15 -> K=hh*16+i,
  // contiguous K order, same as the B tiles.
  v16h aQ[2], aF[2];
#pragma unroll
  for (int t = 0; t < 2; ++t) {
    const float* qrow = q + (((size_t)b * cL + l0 + t * 16 + nIdx) * cH + h) * cE;
    const float* krow = k + (((size_t)b * cL + l0 + t * 16 + nIdx) * cH + h) * cE;
    const float4* q4 = (const float4*)(qrow);
    const float4* k4 = (const float4*)(krow);
    float qf[16], kf[16];
    *(float4*)&qf[0]  = q4[hh * 4 + 0]; *(float4*)&qf[4]  = q4[hh * 4 + 1];
    *(float4*)&qf[8]  = q4[hh * 4 + 2]; *(float4*)&qf[12] = q4[hh * 4 + 3];
    *(float4*)&kf[0]  = k4[hh * 4 + 0]; *(float4*)&kf[4]  = k4[hh * 4 + 1];
    *(float4*)&kf[8]  = k4[hh * 4 + 2]; *(float4*)&kf[12] = k4[hh * 4 + 3];
#pragma unroll
    for (int i = 0; i < 16; ++i) {
      aQ[t][i] = (_Float16)qf[i];
      aF[t][i] = (_Float16)(fabsf(qf[i]) + fabsf(kf[i]));
    }
  }

  // ---------------- running state ----------------
  v8f oA[2] = {{}, {}};           // d = 0..15 accumulators (C layout)
  v8f oB[2] = {{}, {}};           // d = 16..31 accumulators
  float mrow[2][8], lrow[2][8];   // per-row (M = hh*8+j) running max / sum
#pragma unroll
  for (int t = 0; t < 2; ++t)
#pragma unroll
    for (int j = 0; j < 8; ++j) { mrow[t][j] = -INFINITY; lrow[t][j] = 0.0f; }

  const v8f zero8 = {};

  // Raw LDS byte address of this lane's V-stage row (flat -> AS3 -> int)
  const unsigned vldRowLds =
      (unsigned)(unsigned long long)(lds_f*)&Vld[lane * 33];

  // ---------------- main loop over s in chunks of 32 ----------------
  for (int sBase = 0; sBase < cS; sBase += 32) {
    // ---- async V stage: row (sBase+lane), 8 x 16B segments, mem -> LDS ----
    {
      const float* vrow = v + (((size_t)b * cL + sBase + lane) * cH + h) * cD;
      unsigned long long ga = (unsigned long long)(uintptr_t)vrow;
#pragma unroll
      for (int i = 0; i < 8; ++i) {
        unsigned long long gai = ga + (unsigned)(i * 16);
        unsigned lai = vldRowLds + (unsigned)(i * 16);
        asm volatile("global_load_async_to_lds_b128 %0, %1, off"
                     :: "v"(lai), "v"(gai)
                     : "memory");
      }
    }

    // B tiles for keys & w: lane(N=nIdx, hh) holds K-dim e = hh*16 + 0..15
    // of column N (= row sBase+u*16+N of keys/w) -> contiguous 16 floats.
    const float* kr0 = k + (((size_t)b * cL + sBase + nIdx) * cH + h) * cE + hh * 16;
    const float* kr1 = kr0 + (size_t)16 * cH * cE;
    const float* wr0 = w + (((size_t)(sBase + nIdx)) * cH + h) * cE + hh * 16;
    const float* wr1 = wr0 + (size_t)16 * cH * cE;

    // CDNA5 prefetch hint for next chunk (global_prefetch_b8)
    __builtin_prefetch(kr0 + (size_t)32 * cH * cE, 0, 1);
    __builtin_prefetch(wr0 + (size_t)32 * cH * cE, 0, 1);

    v16h bk0, bk1, bw0, bw1;
#pragma unroll
    for (int i = 0; i < 4; ++i) {
      float4 a0 = ((const float4*)kr0)[i];
      float4 a1 = ((const float4*)kr1)[i];
      float4 c0 = ((const float4*)wr0)[i];
      float4 c1 = ((const float4*)wr1)[i];
      bk0[4 * i + 0] = (_Float16)a0.x; bk0[4 * i + 1] = (_Float16)a0.y;
      bk0[4 * i + 2] = (_Float16)a0.z; bk0[4 * i + 3] = (_Float16)a0.w;
      bk1[4 * i + 0] = (_Float16)a1.x; bk1[4 * i + 1] = (_Float16)a1.y;
      bk1[4 * i + 2] = (_Float16)a1.z; bk1[4 * i + 3] = (_Float16)a1.w;
      bw0[4 * i + 0] = (_Float16)c0.x; bw0[4 * i + 1] = (_Float16)c0.y;
      bw0[4 * i + 2] = (_Float16)c0.z; bw0[4 * i + 3] = (_Float16)c0.w;
      bw1[4 * i + 0] = (_Float16)c1.x; bw1[4 * i + 1] = (_Float16)c1.y;
      bw1[4 * i + 2] = (_Float16)c1.z; bw1[4 * i + 3] = (_Float16)c1.w;
    }

    // ---- score GEMMs + gate + online softmax, both l-tiles ----
    // (async V copy runs concurrently with all of this)
#pragma unroll
    for (int t = 0; t < 2; ++t) {
      v8f sc0 = WMMA16F(aQ[t], bk0, zero8);
      v8f sc1 = WMMA16F(aQ[t], bk1, zero8);
      v8f g20 = WMMA16F(aF[t], bw0, zero8);
      v8f g21 = WMMA16F(aF[t], bw1, zero8);

      // C layout: row M = hh*8+j, col N = nIdx
#pragma unroll
      for (int j = 0; j < 8; ++j) {
        float x0 = scale * clip1000(sc0[j]);
        float x1 = scale * clip1000(sc1[j]);
        float y0 = scale * clip1000(g20[j]);
        float y1 = scale * clip1000(g21[j]);
        float gg0 = tanhf(x0) * (1.0f / (1.0f + __expf(-y0)));
        float gg1 = tanhf(x1) * (1.0f / (1.0f + __expf(-y1)));

        // row max over the 16 lanes of this half-wave (both s sub-tiles)
        float rm = fmaxf(gg0, gg1);
        rm = fmaxf(rm, __shfl_xor(rm, 8, 16));
        rm = fmaxf(rm, __shfl_xor(rm, 4, 16));
        rm = fmaxf(rm, __shfl_xor(rm, 2, 16));
        rm = fmaxf(rm, __shfl_xor(rm, 1, 16));

        float mNew = fmaxf(mrow[t][j], rm);
        float corr = __expf(mrow[t][j] - mNew);
        mrow[t][j] = mNew;
        float p0 = __expf(gg0 - mNew);
        float p1 = __expf(gg1 - mNew);

        float rs = p0 + p1;
        rs += __shfl_xor(rs, 8, 16);
        rs += __shfl_xor(rs, 4, 16);
        rs += __shfl_xor(rs, 2, 16);
        rs += __shfl_xor(rs, 1, 16);

        lrow[t][j] = lrow[t][j] * corr + rs;
        oA[t][j] *= corr;
        oB[t][j] *= corr;

        Pld[t][(hh * 8 + j) * 32 + nIdx]      = p0;
        Pld[t][(hh * 8 + j) * 32 + 16 + nIdx] = p1;
      }
    }

    // Drain the async V copy, then make P/V visible for the LDS reads.
    asm volatile("s_wait_asynccnt 0" ::: "memory");
    __syncthreads();

    // V as B tiles (shared by both l-tiles): lane(N,hh) -> column d=N(+16),
    // rows K = hh*16 + 0..15
    v16h bv0, bv1;
#pragma unroll
    for (int i = 0; i < 16; ++i) {
      bv0[i] = (_Float16)Vld[(hh * 16 + i) * 33 + nIdx];
      bv1[i] = (_Float16)Vld[(hh * 16 + i) * 33 + 16 + nIdx];
    }

    // ---- P*V: 4x v_wmma accumulating directly into O ----
#pragma unroll
    for (int t = 0; t < 2; ++t) {
      v16h pa;
#pragma unroll
      for (int i = 0; i < 8; ++i) {
        pa[i]     = (_Float16)Pld[t][nIdx * 32 + hh * 16 + i];      // K = hh*16+i
        pa[8 + i] = (_Float16)Pld[t][nIdx * 32 + hh * 16 + 8 + i];  // K = hh*16+8+i
      }
      oA[t] = WMMA16F(pa, bv0, oA[t]);
      oB[t] = WMMA16F(pa, bv1, oB[t]);
    }

    __syncthreads();  // protect Pld/Vld before next chunk overwrites
  }

  // ---------------- epilogue: normalize and store ----------------
#pragma unroll
  for (int t = 0; t < 2; ++t) {
#pragma unroll
    for (int j = 0; j < 8; ++j) {
      const int row = t * 16 + hh * 8 + j;
      const float inv = 1.0f / lrow[t][j];
      float* orow = out + (((size_t)b * cL + l0 + row) * cH + h) * cD;
      orow[nIdx]      = oA[t][j] * inv;
      orow[16 + nIdx] = oB[t][j] * inv;
    }
  }
}

extern "C" void kernel_launch(void* const* d_in, const int* in_sizes, int n_in,
                              void* d_out, int out_size, void* d_ws, size_t ws_size,
                              hipStream_t stream) {
  const float* q = (const float*)d_in[0];   // [B,L,H,E]
  const float* k = (const float*)d_in[1];   // [B,L,H,E]
  const float* v = (const float*)d_in[2];   // [B,L,H,D]
  const float* w = (const float*)d_in[3];   // [1,L,H,E]
  // d_in[4] = attn_mask (unused by reference forward)
  float* out = (float*)d_out;               // [B,L,H,D]

  const int nBlocks = cB * cH * (cL / 32);  // 2048 single-wave workgroups
  gated_attn_kernel<<<nBlocks, 32, 0, stream>>>(q, k, v, w, out);
}
